// AttentionLSTMDecoder_54485955117309
// MI455X (gfx1250) — compile-verified, matrix operand order checked by reading, not run on hardware
//
#include <hip/hip_runtime.h>
#include <hip/hip_bf16.h>

// ---------------- types for CDNA5 WMMA (wave32) ----------------
typedef __bf16 bf16;
typedef __attribute__((ext_vector_type(16))) __bf16        v16bf;
typedef __attribute__((ext_vector_type(8)))  float         v8f;
typedef __attribute__((ext_vector_type(8)))  unsigned int  v8u;

// Model dims (fixed by the reference)
#define VOCAB 8000
#define EDIM  512
#define HE    1024
#define HD    1024
#define ADIM  256
#define NL    4
#define BATCH 64
#define SLEN  128

// Fragment gather for the ISA 16-bit 16x32 layout: lane m holds K {0..7,16..23},
// lane m+16 holds K {8..15,24..31}. The {0,2,4,6} dword offsets are contiguous
// 16B runs -> compiler merges them into global_load_b128 pairs.
__device__ __forceinline__ void load_frag(const bf16* __restrict__ row,
                                          int kb, v8u& u) {
#pragma unroll
  for (int j = 0; j < 8; ++j) {
    const int k = kb + ((j < 4) ? 2 * j : 16 + 2 * (j - 4));
    u[j] = *(const unsigned int*)(row + k);
  }
}
__device__ __forceinline__ void load_a4(const bf16* __restrict__ Xrow,
                                        size_t ldx, int kb, v8u ua[4]) {
#pragma unroll
  for (int m = 0; m < 4; ++m)
    load_frag(Xrow + (size_t)(16 * m) * ldx, kb, ua[m]);
}

// M-blocked, software-pipelined K loop: one wave owns a 64x16 output strip
// (4 WMMA accumulators sharing one B fragment), so each weight strip is read
// from L2 by exactly one wave. Fragments for chunk kb+32 are issued before
// the 4 WMMAs on chunk kb.
__device__ __forceinline__ void gemm_accum4(v8f acc[4],
                                            const bf16* __restrict__ Xrow, size_t ldx,
                                            const bf16* __restrict__ Wrow,
                                            int K) {
  v8u a0[4], a1[4], b0, b1;
  load_a4(Xrow, ldx, 0, a0);
  load_frag(Wrow, 0, b0);
  int kb = 0;
  for (; kb + 32 < K; kb += 32) {
    load_a4(Xrow, ldx, kb + 32, a1);               // next chunk in flight
    load_frag(Wrow, kb + 32, b1);
    __builtin_prefetch(Wrow + kb + 1024, 0, 1);    // global_prefetch_b8
#pragma unroll
    for (int m = 0; m < 4; ++m)
      acc[m] = __builtin_amdgcn_wmma_f32_16x16x32_bf16(
          false, __builtin_bit_cast(v16bf, a0[m]),
          false, __builtin_bit_cast(v16bf, b0),
          (short)0, acc[m], false, false);
#pragma unroll
    for (int m = 0; m < 4; ++m) a0[m] = a1[m];
    b0 = b1;
  }
#pragma unroll
  for (int m = 0; m < 4; ++m)
    acc[m] = __builtin_amdgcn_wmma_f32_16x16x32_bf16(
        false, __builtin_bit_cast(v16bf, a0[m]),
        false, __builtin_bit_cast(v16bf, b0),
        (short)0, acc[m], false, false);
}

// =====================================================================
// Generic bf16 GEMM:  C[M,N] = X[M,K] W[N,K]^T (+ X2[M,K2] W2[N,K2]^T)
//                              (+bias | +Cin accumulate)
// Requires M % 64 == 0 (true for all uses: 64 or 8192).
// Block = 256 threads = 8 waves; wave w covers cols [base+16w, base+16w+16),
// rows [64*blockIdx.y, 64*blockIdx.y+64). Tile guards are wave-uniform ->
// EXEC all-ones at WMMA (ISA 7.12 requirement).
// =====================================================================
__global__ void gemm_bf16_wmma(const bf16* __restrict__ X, int ldx,
                               const bf16* __restrict__ W, int ldw,
                               const bf16* __restrict__ X2, int ldx2,
                               const bf16* __restrict__ W2, int ldw2, int K2,
                               float* __restrict__ C, int ldc,
                               const float* __restrict__ bias,
                               const float* __restrict__ Cin,
                               int N, int K) {
  const int tid  = threadIdx.x;
  const int lane = tid & 31;
  const int wave = tid >> 5;
  const int n0 = (blockIdx.x * 8 + wave) * 16;
  const int m0 = blockIdx.y * 64;
  if (n0 >= N) return;                      // uniform per wave

  const int half = lane >> 4;
  const int l15  = lane & 15;

  v8f acc[4];
  if (Cin) {
#pragma unroll
    for (int m = 0; m < 4; ++m)
#pragma unroll
      for (int r = 0; r < 8; ++r)
        acc[m][r] = Cin[(size_t)(m0 + 16 * m + r + 8 * half) * ldc + n0 + l15];
  } else {
    const float bv = bias ? bias[n0 + l15] : 0.f;
#pragma unroll
    for (int m = 0; m < 4; ++m)
#pragma unroll
      for (int r = 0; r < 8; ++r) acc[m][r] = bv;
  }

  const bf16* Xrow = X + (size_t)(m0 + l15) * ldx + 8 * half;
  const bf16* Wrow = W + (size_t)(n0 + l15) * ldw + 8 * half;
  gemm_accum4(acc, Xrow, (size_t)ldx, Wrow, K);

  if (X2) {
    const bf16* X2row = X2 + (size_t)(m0 + l15) * ldx2 + 8 * half;
    const bf16* W2row = W2 + (size_t)(n0 + l15) * ldw2 + 8 * half;
    gemm_accum4(acc, X2row, (size_t)ldx2, W2row, K2);
  }

#pragma unroll
  for (int m = 0; m < 4; ++m)
#pragma unroll
    for (int r = 0; r < 8; ++r)
      C[(size_t)(m0 + 16 * m + r + 8 * half) * ldc + n0 + l15] = acc[m][r];
}

// ---------------- block reduction helper (blockDim == 256) ----------------
__device__ __forceinline__ float block_sum256(float v, float* sh) {
  const int tid = threadIdx.x;
  sh[tid] = v; __syncthreads();
  for (int s = 128; s > 0; s >>= 1) {
    if (tid < s) sh[tid] += sh[tid + s];
    __syncthreads();
  }
  const float r = sh[0]; __syncthreads();
  return r;
}

__device__ __forceinline__ float sigm(float x) { return 1.f / (1.f + expf(-x)); }

// ---------------- embedding lookup + LayerNorm -> bf16 x[:, 0:512] --------
__global__ void embed_ln_kernel(const int* __restrict__ tok,
                                const float* __restrict__ emb,
                                const float* __restrict__ g,
                                const float* __restrict__ b,
                                bf16* __restrict__ xbuf) {
  __shared__ float sh[256];
  const int bb = blockIdx.x, tid = threadIdx.x;
  const int t = tok[bb];
  const float v0 = emb[(size_t)t * EDIM + tid];
  const float v1 = emb[(size_t)t * EDIM + 256 + tid];
  const float mean = block_sum256(v0 + v1, sh) * (1.f / EDIM);
  const float d0 = v0 - mean, d1 = v1 - mean;
  const float var = block_sum256(d0 * d0 + d1 * d1, sh) * (1.f / EDIM);
  const float inv = rsqrtf(var + 1e-5f);
  xbuf[(size_t)bb * (EDIM + HE) + tid]       = (bf16)(d0 * inv * g[tid] + b[tid]);
  xbuf[(size_t)bb * (EDIM + HE) + 256 + tid] = (bf16)(d1 * inv * g[256 + tid] + b[256 + tid]);
}

// ---------------- attention: scores, masked softmax, context --------------
__global__ void attention_kernel(const float* __restrict__ enc_proj, // [B,S,A]
                                 const float* __restrict__ dp,       // [B,A]
                                 const float* __restrict__ enc,      // [B,S,HE]
                                 const float* __restrict__ att_v,
                                 const float* __restrict__ att_vb,
                                 const int*   __restrict__ src_len,
                                 bf16* __restrict__ xbuf) {          // ctx -> x[:,512:1536]
  __shared__ float sh[256];
  __shared__ float sdp[ADIM];
  __shared__ float sw[SLEN];
  const int bb = blockIdx.x, tid = threadIdx.x;
  sdp[tid] = dp[(size_t)bb * ADIM + tid];
  __syncthreads();

  const int s = tid & (SLEN - 1), hf = tid >> 7;       // 2 threads per source pos
  const float* ep = enc_proj + ((size_t)bb * SLEN + s) * ADIM + hf * 128;
  float part = 0.f;
#pragma unroll 4
  for (int a = 0; a < 128; ++a)
    part += tanhf(ep[a] + sdp[hf * 128 + a]) * att_v[hf * 128 + a];
  sh[tid] = part; __syncthreads();

  const int len = src_len[bb];
  if (tid < SLEN) {
    float sc = sh[tid] + sh[tid + 128] + att_vb[0];
    sw[tid] = (tid < len) ? sc : -1e9f;
  }
  __syncthreads();

  // max
  sh[tid] = (tid < SLEN) ? sw[tid] : -3.0e38f; __syncthreads();
  for (int st = 128; st > 0; st >>= 1) { if (tid < st) sh[tid] = fmaxf(sh[tid], sh[tid + st]); __syncthreads(); }
  const float mx = sh[0]; __syncthreads();
  if (tid < SLEN) sw[tid] = expf(sw[tid] - mx);
  __syncthreads();
  // sum
  sh[tid] = (tid < SLEN) ? sw[tid] : 0.f; __syncthreads();
  for (int st = 128; st > 0; st >>= 1) { if (tid < st) sh[tid] += sh[tid + st]; __syncthreads(); }
  const float rden = 1.f / sh[0]; __syncthreads();
  if (tid < SLEN) sw[tid] *= rden;
  __syncthreads();

  // ctx[b,e] = sum_s w[s]*enc[b,s,e] ; 4 coalesced columns per thread
  float acc[4] = {0.f, 0.f, 0.f, 0.f};
  for (int ss = 0; ss < SLEN; ++ss) {
    const float w = sw[ss];
    const float* er = enc + ((size_t)bb * SLEN + ss) * HE;
#pragma unroll
    for (int j = 0; j < 4; ++j) acc[j] += w * er[tid + 256 * j];
  }
#pragma unroll
  for (int j = 0; j < 4; ++j)
    xbuf[(size_t)bb * (EDIM + HE) + EDIM + tid + 256 * j] = (bf16)acc[j];
}

// ---------------- LSTM cell (i,f,g,o) + LayerNorm -> bf16 h ---------------
__global__ void lstm_cell_kernel(const float* __restrict__ gates, // [B,4*HD]
                                 float* __restrict__ c_st,        // [B,HD]
                                 bf16*  __restrict__ h_out,       // [B,HD] bf16
                                 const float* __restrict__ g,
                                 const float* __restrict__ bb) {
  __shared__ float sh[256];
  const int b = blockIdx.x, tid = threadIdx.x;
  float hraw[4]; float s = 0.f;
#pragma unroll
  for (int j = 0; j < 4; ++j) {
    const int u = tid + 256 * j;
    const float ig = gates[(size_t)b * 4 * HD + u];
    const float fg = gates[(size_t)b * 4 * HD + HD + u];
    const float gg = gates[(size_t)b * 4 * HD + 2 * HD + u];
    const float og = gates[(size_t)b * 4 * HD + 3 * HD + u];
    const float c = sigm(fg) * c_st[(size_t)b * HD + u] + sigm(ig) * tanhf(gg);
    c_st[(size_t)b * HD + u] = c;
    const float h = sigm(og) * tanhf(c);
    hraw[j] = h; s += h;
  }
  const float mean = block_sum256(s, sh) * (1.f / HD);
  float vs = 0.f;
#pragma unroll
  for (int j = 0; j < 4; ++j) { const float d = hraw[j] - mean; vs += d * d; }
  const float var = block_sum256(vs, sh) * (1.f / HD);
  const float inv = rsqrtf(var + 1e-5f);
#pragma unroll
  for (int j = 0; j < 4; ++j) {
    const int u = tid + 256 * j;
    h_out[(size_t)b * HD + u] = (bf16)((hraw[j] - mean) * inv * g[u] + bb[u]);
  }
}

// ---------------- gelu(z) + top residual -> bf16 --------------------------
__global__ void gelu_res_kernel(const float* __restrict__ z,
                                const bf16* __restrict__ top,
                                bf16* __restrict__ out) {
  const int b = blockIdx.x, tid = threadIdx.x;
#pragma unroll
  for (int j = 0; j < 4; ++j) {
    const int u = tid + 256 * j;
    const float x = z[(size_t)b * HD + u];
    const float ge = 0.5f * x * (1.f + erff(x * 0.70710678118654752f));
    out[(size_t)b * HD + u] = (bf16)(ge + (float)top[(size_t)b * HD + u]);
  }
}

// ---------------- argmax over vocab (first-max tiebreak) ------------------
__global__ void argmax_kernel(const float* __restrict__ logits, int ld,
                              int* __restrict__ tok) {
  __shared__ float smax[256];
  __shared__ int   sidx[256];
  const int b = blockIdx.x, tid = threadIdx.x;
  const float* row = logits + (size_t)b * ld;
  float best = -3.0e38f; int bi = 0;
  for (int v = tid; v < VOCAB; v += 256) {
    const float x = row[v];
    if (x > best) { best = x; bi = v; }
  }
  smax[tid] = best; sidx[tid] = bi; __syncthreads();
  for (int s = 128; s > 0; s >>= 1) {
    if (tid < s) {
      if (smax[tid + s] > smax[tid] ||
          (smax[tid + s] == smax[tid] && sidx[tid + s] < sidx[tid])) {
        smax[tid] = smax[tid + s]; sidx[tid] = sidx[tid + s];
      }
    }
    __syncthreads();
  }
  if (tid == 0) tok[b] = sidx[0];
}

// ---------------- small utility kernels -----------------------------------
__global__ void f32_to_bf16_kernel(const float* __restrict__ s,
                                   bf16* __restrict__ d, int n) {
  const int i = blockIdx.x * 256 + threadIdx.x;
  if (i < n) d[i] = (bf16)s[i];
}
__global__ void vec_add_kernel(const float* __restrict__ a,
                               const float* __restrict__ b,
                               float* __restrict__ o, int n) {
  const int i = blockIdx.x * 256 + threadIdx.x;
  if (i < n) o[i] = a[i] + b[i];
}
// cat(hidden[2], hidden[3]) over feature dim -> bf16 [B, 2*HE]
__global__ void pack_final_kernel(const float* __restrict__ hid,
                                  bf16* __restrict__ out) {
  const int i = blockIdx.x * 256 + threadIdx.x;
  if (i >= BATCH * 2 * HE) return;
  const int b = i >> 11, e = i & 2047;
  const int l = (e < HE) ? 2 : 3;
  out[i] = (bf16)hid[(size_t)l * BATCH * HE + (size_t)b * HE + (e & (HE - 1))];
}
// split bridge outputs [B, NL*HD] into per-layer states; token <- SOS(3)
__global__ void init_state_kernel(const float* __restrict__ h0,
                                  const float* __restrict__ c0,
                                  bf16* __restrict__ h_st, float* __restrict__ c_st,
                                  int* __restrict__ tok) {
  const int i = blockIdx.x * 256 + threadIdx.x;
  if (i < BATCH) tok[i] = 3;
  if (i >= NL * BATCH * HD) return;
  const int l = i >> 16, rem = i & 65535, b = rem >> 10, u = rem & 1023;
  h_st[(size_t)l * BATCH * HD + (size_t)b * HD + u] = (bf16)h0[(size_t)b * NL * HD + l * HD + u];
  c_st[(size_t)l * BATCH * HD + (size_t)b * HD + u] = c0[(size_t)b * NL * HD + l * HD + u];
}

// =====================================================================
// Host driver
// =====================================================================
static inline void launch_gemm2(const bf16* X, int ldx, const bf16* W, int ldw,
                                const bf16* X2, int ldx2, const bf16* W2, int ldw2, int K2,
                                float* C, int ldc, const float* bias,
                                const float* Cin, int M, int N, int K,
                                hipStream_t s) {
  dim3 grid((N + 127) / 128, M / 64);           // M % 64 == 0 for all uses
  gemm_bf16_wmma<<<grid, 256, 0, s>>>(X, ldx, W, ldw, X2, ldx2, W2, ldw2, K2,
                                      C, ldc, bias, Cin, N, K);
}
static inline void launch_gemm(const bf16* X, int ldx, const bf16* W, int ldw,
                               float* C, int ldc, const float* bias,
                               const float* Cin, int M, int N, int K,
                               hipStream_t s) {
  launch_gemm2(X, ldx, W, ldw, nullptr, 0, nullptr, 0, 0,
               C, ldc, bias, Cin, M, N, K, s);
}

extern "C" void kernel_launch(void* const* d_in, const int* in_sizes, int n_in,
                              void* d_out, int out_size, void* d_ws, size_t ws_size,
                              hipStream_t stream) {
  (void)in_sizes; (void)n_in; (void)ws_size;
  const float* enc      = (const float*)d_in[0];
  const float* enc_hid  = (const float*)d_in[1];
  const float* enc_cell = (const float*)d_in[2];
  const int*   src_len  = (const int*)d_in[3];
  // d_in[4] = max_length (device scalar); T derived from out_size instead
  const float* emb   = (const float*)d_in[5];
  const float* emb_g = (const float*)d_in[6];
  const float* emb_b = (const float*)d_in[7];
  const float* W_enc = (const float*)d_in[8];   const float* b_enc = (const float*)d_in[9];
  const float* W_dec = (const float*)d_in[10];  const float* b_dec = (const float*)d_in[11];
  const float* att_v = (const float*)d_in[12];  const float* att_vb = (const float*)d_in[13];
  const float *Wih[NL], *Whh[NL], *bih[NL], *bhh[NL];
  for (int l = 0; l < NL; ++l) {
    Wih[l] = (const float*)d_in[14 + l];
    Whh[l] = (const float*)d_in[18 + l];
    bih[l] = (const float*)d_in[22 + l];
    bhh[l] = (const float*)d_in[26 + l];
  }
  const float* ln_g = (const float*)d_in[30];   const float* ln_b = (const float*)d_in[31];
  const float* W_hbr = (const float*)d_in[32];  const float* b_hbr = (const float*)d_in[33];
  const float* W_cbr = (const float*)d_in[34];  const float* b_cbr = (const float*)d_in[35];
  const float* W_cp  = (const float*)d_in[36];  const float* b_cp  = (const float*)d_in[37];
  const float* W_op  = (const float*)d_in[38];  const float* b_op  = (const float*)d_in[39];
  const float* W_fo  = (const float*)d_in[40];  const float* b_fo  = (const float*)d_in[41];
  float* out = (float*)d_out;
  int T = out_size / (BATCH * VOCAB);
  if (T <= 0) T = 64;

  // ---- workspace carve-up (~144 MB total; bf16 weights ~131 MB -> L2 resident)
  char* p = (char*)d_ws;
  auto alloc = [&](size_t bytes) -> char* {
    char* r = p; p += (bytes + 255) & ~(size_t)255; return r;
  };
  auto cvt = [&](const float* src, size_t n) -> bf16* {
    bf16* d = (bf16*)alloc(n * sizeof(bf16));
    f32_to_bf16_kernel<<<(int)((n + 255) / 256), 256, 0, stream>>>(src, d, (int)n);
    return d;
  };

  // one-time (per call) bf16 weight conversion
  bf16* wenc = cvt(W_enc, (size_t)ADIM * HE);
  bf16* wdec = cvt(W_dec, (size_t)ADIM * HD);
  bf16 *wih[NL], *whh[NL];
  const int kin[NL] = {EDIM + HE, HD, HD, HD};
  for (int l = 0; l < NL; ++l) {
    wih[l] = cvt(Wih[l], (size_t)4 * HD * kin[l]);
    whh[l] = cvt(Whh[l], (size_t)4 * HD * HD);
  }
  bf16* whbr = cvt(W_hbr, (size_t)NL * HD * 2 * HE);
  bf16* wcbr = cvt(W_cbr, (size_t)NL * HD * 2 * HE);
  bf16* wcp  = cvt(W_cp,  (size_t)HD * HE);
  bf16* wop  = cvt(W_op,  (size_t)HD * 2 * HD);
  bf16* wfo  = cvt(W_fo,  (size_t)VOCAB * HD);
  bf16* encbf = cvt(enc,  (size_t)BATCH * SLEN * HE);

  // combined LSTM biases
  float* bsum = (float*)alloc((size_t)NL * 4 * HD * sizeof(float));
  for (int l = 0; l < NL; ++l)
    vec_add_kernel<<<(4 * HD + 255) / 256, 256, 0, stream>>>(bih[l], bhh[l], bsum + (size_t)l * 4 * HD, 4 * HD);

  // activations / state
  float* enc_proj = (float*)alloc((size_t)BATCH * SLEN * ADIM * sizeof(float));
  float* dp    = (float*)alloc((size_t)BATCH * ADIM * sizeof(float));
  float* gates = (float*)alloc((size_t)BATCH * 4 * HD * sizeof(float));
  float* c_st  = (float*)alloc((size_t)NL * BATCH * HD * sizeof(float));
  float* h0    = (float*)alloc((size_t)BATCH * NL * HD * sizeof(float));
  float* c0    = (float*)alloc((size_t)BATCH * NL * HD * sizeof(float));
  float* pcf   = (float*)alloc((size_t)BATCH * HD * sizeof(float));
  float* zbuf  = (float*)alloc((size_t)BATCH * HD * sizeof(float));
  bf16* h_st  = (bf16*)alloc((size_t)NL * BATCH * HD * sizeof(bf16));
  bf16* xbuf0 = (bf16*)alloc((size_t)BATCH * (EDIM + HE) * sizeof(bf16));
  bf16* pcbf  = (bf16*)alloc((size_t)BATCH * HD * sizeof(bf16));
  bf16* outbf = (bf16*)alloc((size_t)BATCH * HD * sizeof(bf16));
  bf16* finalh = (bf16*)alloc((size_t)BATCH * 2 * HE * sizeof(bf16));
  bf16* finalc = (bf16*)alloc((size_t)BATCH * 2 * HE * sizeof(bf16));
  int* tok = (int*)alloc(BATCH * sizeof(int));

  // ---- step-invariant precompute
  // enc_proj[B*S, A] = enc_bf16 @ W_enc^T + b_enc
  launch_gemm(encbf, HE, wenc, HE, enc_proj, ADIM, b_enc, nullptr,
              BATCH * SLEN, ADIM, HE, stream);
  pack_final_kernel<<<(BATCH * 2 * HE + 255) / 256, 256, 0, stream>>>(enc_hid, finalh);
  pack_final_kernel<<<(BATCH * 2 * HE + 255) / 256, 256, 0, stream>>>(enc_cell, finalc);
  launch_gemm(finalh, 2 * HE, whbr, 2 * HE, h0, NL * HD, b_hbr, nullptr,
              BATCH, NL * HD, 2 * HE, stream);
  launch_gemm(finalc, 2 * HE, wcbr, 2 * HE, c0, NL * HD, b_cbr, nullptr,
              BATCH, NL * HD, 2 * HE, stream);
  init_state_kernel<<<(NL * BATCH * HD + 255) / 256, 256, 0, stream>>>(h0, c0, h_st, c_st, tok);

  // ---- sequential greedy decode
  const size_t lsz = (size_t)BATCH * HD;   // per-layer state slab
  for (int t = 0; t < T; ++t) {
    embed_ln_kernel<<<BATCH, 256, 0, stream>>>(tok, emb, emb_g, emb_b, xbuf0);
    // dp = h[3] @ W_dec^T + b_dec
    launch_gemm(h_st + 3 * lsz, HD, wdec, HD, dp, ADIM, b_dec, nullptr,
                BATCH, ADIM, HD, stream);
    attention_kernel<<<BATCH, 256, 0, stream>>>(enc_proj, dp, enc, att_v, att_vb,
                                                src_len, xbuf0);
    for (int l = 0; l < NL; ++l) {
      const bf16* X = (l == 0) ? xbuf0 : h_st + (size_t)(l - 1) * lsz;
      const int ldx = (l == 0) ? (EDIM + HE) : HD;
      // fused: gates = x @ Wih^T + h_prev @ Whh^T + (bih+bhh), one pass
      launch_gemm2(X, ldx, wih[l], kin[l],
                   h_st + (size_t)l * lsz, HD, whh[l], HD, HD,
                   gates, 4 * HD, bsum + (size_t)l * 4 * HD, nullptr,
                   BATCH, 4 * HD, kin[l], stream);
      lstm_cell_kernel<<<BATCH, 256, 0, stream>>>(gates, c_st + (size_t)l * lsz,
                                                  h_st + (size_t)l * lsz,
                                                  ln_g + (size_t)l * HD,
                                                  ln_b + (size_t)l * HD);
    }
    // pc = ctx @ W_cp^T + b_cp  (ctx lives in xbuf0[:, 512:1536])
    launch_gemm(xbuf0 + EDIM, EDIM + HE, wcp, HE, pcf, HD, b_cp, nullptr,
                BATCH, HD, HE, stream);
    f32_to_bf16_kernel<<<((int)lsz + 255) / 256, 256, 0, stream>>>(pcf, pcbf, (int)lsz);
    // fused: z = top @ W_op[:,:HD]^T + pc @ W_op[:,HD:]^T + b_op, one pass
    launch_gemm2(h_st + 3 * lsz, HD, wop, 2 * HD,
                 pcbf, HD, wop + HD, 2 * HD, HD,
                 zbuf, HD, b_op, nullptr, BATCH, HD, HD, stream);
    gelu_res_kernel<<<BATCH, 256, 0, stream>>>(zbuf, h_st + 3 * lsz, outbf);
    // logits -> d_out[b, t, :] directly (row stride T*VOCAB)
    launch_gemm(outbf, HD, wfo, HD, out + (size_t)t * VOCAB, T * VOCAB, b_fo,
                nullptr, BATCH, VOCAB, HD, stream);
    argmax_kernel<<<BATCH, 256, 0, stream>>>(out + (size_t)t * VOCAB, T * VOCAB, tok);
  }
}
